// ElmanRnn_33011118637553
// MI455X (gfx1250) — matrix operand order repaired and checked by reading, not compiled, not run on hardware
//
#include <hip/hip_runtime.h>
#include <math.h>

// ---------------------------------------------------------------------------
// Elman RNN (B=32, S=1024, D=1024, L=2) persistent-kernel implementation.
// Weights stay resident in LDS as bf16 (B-operand layout); hidden state is
// parity-double-buffered bf16 in workspace; one grid-wide barrier per step;
// layer-0(t) and layer-1(t-1) run concurrently (software pipelining).
// If workspace is large enough, x is pre-converted to bf16 once so the hot
// loop is pure {2x ds_load_b128 + 2x global_load_b128 + v_wmma} per K-chunk.
// ---------------------------------------------------------------------------

typedef __attribute__((ext_vector_type(16))) __bf16 v16bf;
typedef __attribute__((ext_vector_type(8)))  __bf16 v8bf;
typedef __attribute__((ext_vector_type(4)))  __bf16 v4bf;
typedef __attribute__((ext_vector_type(8)))  float  v8f;
typedef __attribute__((ext_vector_type(4)))  float  v4f;

#define RNN_B   32
#define RNN_S   1024
#define RNN_D   1024
#define NWG     64          // one N-tile (16 cols) per workgroup
#define TPB     128         // 4 waves: {layer0,layer1} x {m_tile0,m_tile1}
#define KTOT    2048        // fused K = [X | H] against [W ; U]
#define KPAD    2056        // 2048 + 8 elements: lane stride 4112B -> 4 banks apart
#define LDS_ELEMS (2 * 16 * KPAD)   // 2 layers x 16 cols x padded K  (~128.5 KB)

// ---- A operand: 16x32 bf16 tile per ISA 7.12.2 ----------------------------
// lane L: M = L&15, half = L>>4; elements e<8 -> K = kb + half*8 + e,
// e>=8 -> K = kb + 16 + half*8 + (e-8).  Two contiguous 16B groups.
__device__ __forceinline__ v16bf ldA_bf16(const __bf16* __restrict__ row,
                                          int kb, int hf) {
  v8bf g0 = *(const v8bf*)(row + kb + hf * 8);
  v8bf g1 = *(const v8bf*)(row + kb + 16 + hf * 8);
  v16bf a;
#pragma unroll
  for (int i = 0; i < 8; ++i) { a[i] = g0[i]; a[i + 8] = g1[i]; }
  return a;
}

__device__ __forceinline__ v16bf ldA_f32(const float* __restrict__ row,
                                         int kb, int hf) {
  v4f f0 = *(const v4f*)(row + kb + hf * 8);
  v4f f1 = *(const v4f*)(row + kb + hf * 8 + 4);
  v4f f2 = *(const v4f*)(row + kb + 16 + hf * 8);
  v4f f3 = *(const v4f*)(row + kb + 16 + hf * 8 + 4);
  v16bf a;
#pragma unroll
  for (int i = 0; i < 4; ++i) {
    a[i]      = (__bf16)f0[i];
    a[i + 4]  = (__bf16)f1[i];
    a[i + 8]  = (__bf16)f2[i];
    a[i + 12] = (__bf16)f3[i];
  }
  return a;
}

// ---- B operand: 32x16 bf16, lane L holds column N=L&15, K = half*8 + e (e<8), 16 + half*8 + (e-8) (e>=8) ---
// p already includes (col*KPAD + half*8); two 8-bf16 groups 16 apart = two b128 ds loads.
__device__ __forceinline__ v16bf ldB(const __bf16* __restrict__ p) {
  v8bf g0 = *(const v8bf*)(p);
  v8bf g1 = *(const v8bf*)(p + 16);
  v16bf b;
#pragma unroll
  for (int i = 0; i < 8; ++i) { b[i] = g0[i]; b[i + 8] = g1[i]; }
  return b;
}

__device__ __forceinline__ v8f wmma_bf16(v16bf a, v16bf b, v8f c) {
  return __builtin_amdgcn_wmma_f32_16x16x32_bf16(
      /*neg_a=*/false, a, /*neg_b=*/false, b,
      /*c_mod=*/(short)0, c, /*reuse_a=*/false, /*reuse_b=*/false);
}

extern __shared__ __bf16 g_lds[];

__global__ void __launch_bounds__(TPB)
rnn_persistent(const float* __restrict__ x,      // [B,S,D] fp32
               const __bf16* __restrict__ xbf,   // [B,S,D] bf16 (optional)
               const int use_xbf,
               const float* __restrict__ Wh,     // [2,D,D] fp32
               const float* __restrict__ Uh,     // [2,D,D] fp32
               const float* __restrict__ bias,   // [2,D]   fp32
               float* __restrict__ out,          // [B,S,D] fp32
               __bf16* __restrict__ h0buf,       // [2][B][D] bf16 (parity)
               __bf16* __restrict__ h1buf,       // [2][B][D] bf16 (parity)
               unsigned* __restrict__ bar) {
  const int wg     = blockIdx.x;        // n_tile = wg (16 output columns)
  const int tid    = threadIdx.x;
  const int wave   = tid >> 5;
  const int lane   = tid & 31;
  const int layer  = wave >> 1;         // waves 0-1: layer0, waves 2-3: layer1
  const int m_tile = wave & 1;
  const int col    = lane & 15;
  const int hf     = lane >> 4;

  // ------- one-time: convert this WG's weight slice fp32 -> bf16 into LDS --
  // LDS layout: [layer][col 0..15][K padded KPAD];  K<1024 -> W, else U.
  for (int l = 0; l < 2; ++l) {
    const float* Wsrc = Wh + (size_t)l * RNN_D * RNN_D;
    const float* Usrc = Uh + (size_t)l * RNN_D * RNN_D;
    for (int idx = tid; idx < 16 * KTOT; idx += TPB) {
      int c = idx & 15;
      int k = idx >> 4;
      int gcol = wg * 16 + c;
      float v = (k < RNN_D) ? Wsrc[(size_t)k * RNN_D + gcol]
                            : Usrc[(size_t)(k - RNN_D) * RNN_D + gcol];
      g_lds[(size_t)l * 16 * KPAD + (size_t)c * KPAD + k] = (__bf16)v;
    }
  }
  __syncthreads();

  const __bf16* bslice = g_lds + (size_t)layer * 16 * KPAD
                               + (size_t)col * KPAD + hf * 8;
  const float  biasv = bias[layer * RNN_D + wg * 16 + col];
  const int    brow  = m_tile * 16 + col;          // A-matrix batch row for this lane

  // ------- pipelined time loop: phase t runs layer0(t) and layer1(t-1) -----
  for (int t = 0; t <= RNN_S; ++t) {
    const bool active = (layer == 0) ? (t < RNN_S) : (t >= 1);
    if (active) {
      const int te = (layer == 0) ? t : t - 1;     // effective time index
      v8f acc = {};

      if (layer == 0) {
        // X part: x[:, te, :]
        if (use_xbf) {
          const __bf16* xr = xbf + (size_t)brow * RNN_S * RNN_D
                                 + (size_t)te * RNN_D;
#pragma unroll 4
          for (int kc = 0; kc < 32; ++kc) {
            v16bf a = ldA_bf16(xr, kc * 32, hf);
            v16bf b = ldB(bslice + kc * 32);
            acc = wmma_bf16(a, b, acc);
          }
        } else {
          const float* xr = x + (size_t)brow * RNN_S * RNN_D
                              + (size_t)te * RNN_D;
#pragma unroll 4
          for (int kc = 0; kc < 32; ++kc) {
            v16bf a = ldA_f32(xr, kc * 32, hf);
            v16bf b = ldB(bslice + kc * 32);
            acc = wmma_bf16(a, b, acc);
          }
        }
        // H part: h0(t-1), parity (te-1)&1 == (te+1)&1
        const __bf16* hp = h0buf + (size_t)((te + 1) & 1) * RNN_B * RNN_D
                                 + (size_t)brow * RNN_D;
#pragma unroll 4
        for (int kc = 0; kc < 32; ++kc) {
          v16bf a = ldA_bf16(hp, kc * 32, hf);
          v16bf b = ldB(bslice + RNN_D + kc * 32);
          acc = wmma_bf16(a, b, acc);
        }
      } else {
        // layer 1: A = [h0(te) | h1(te-1)]
        const __bf16* hc = h0buf + (size_t)(te & 1) * RNN_B * RNN_D
                                 + (size_t)brow * RNN_D;
#pragma unroll 4
        for (int kc = 0; kc < 32; ++kc) {
          v16bf a = ldA_bf16(hc, kc * 32, hf);
          v16bf b = ldB(bslice + kc * 32);
          acc = wmma_bf16(a, b, acc);
        }
        const __bf16* hp = h1buf + (size_t)((te + 1) & 1) * RNN_B * RNN_D
                                 + (size_t)brow * RNN_D;
#pragma unroll 4
        for (int kc = 0; kc < 32; ++kc) {
          v16bf a = ldA_bf16(hp, kc * 32, hf);
          v16bf b = ldB(bslice + RNN_D + kc * 32);
          acc = wmma_bf16(a, b, acc);
        }
      }

      // ------- epilogue: bias + tanh, store state (+ output for layer1) ----
      __bf16* hdst = ((layer == 0) ? h0buf : h1buf)
                     + (size_t)(te & 1) * RNN_B * RNN_D;
      const int nc = wg * 16 + col;
#pragma unroll
      for (int r = 0; r < 8; ++r) {
        float v = tanhf(acc[r] + biasv);          // C layout: M = r + 8*half
        int bg = m_tile * 16 + hf * 8 + r;
        hdst[(size_t)bg * RNN_D + nc] = (__bf16)v;
        if (layer == 1)
          out[(size_t)bg * RNN_S * RNN_D + (size_t)te * RNN_D + nc] = v;
      }

      // ------- prefetch next step's x row (hide HBM/L2 miss behind barrier)
      if (layer == 0 && te + 1 < RNN_S) {
        if (use_xbf) {
          const __bf16* xn = xbf + (size_t)brow * RNN_S * RNN_D
                                 + (size_t)(te + 1) * RNN_D + hf * 512;
#pragma unroll
          for (int i = 0; i < 4; ++i)
            __builtin_prefetch((const void*)(xn + i * 128), 0, 1);
        } else {
          const float* xn = x + (size_t)brow * RNN_S * RNN_D
                              + (size_t)(te + 1) * RNN_D + hf * 512;
#pragma unroll
          for (int i = 0; i < 4; ++i)
            __builtin_prefetch((const void*)(xn + i * 128), 0, 1);
        }
      }
    }

    // ------- grid-wide step barrier (monotonic counter, release/acquire) ---
    __threadfence();
    __syncthreads();
    if (tid == 0) {
      __hip_atomic_fetch_add(bar, 1u, __ATOMIC_ACQ_REL, __HIP_MEMORY_SCOPE_AGENT);
      const unsigned target = (unsigned)(t + 1) * NWG;
      while (__hip_atomic_load(bar, __ATOMIC_ACQUIRE, __HIP_MEMORY_SCOPE_AGENT)
             < target) {
        __builtin_amdgcn_s_sleep(2);
      }
    }
    __syncthreads();
    __threadfence();
  }
}

// One-time x fp32 -> bf16 conversion (only when workspace is big enough).
__global__ void rnn_cvt_x(const float* __restrict__ x,
                          __bf16* __restrict__ xbf, size_t n4) {
  const size_t stride = (size_t)gridDim.x * blockDim.x;
  for (size_t i = (size_t)blockIdx.x * blockDim.x + threadIdx.x; i < n4;
       i += stride) {
    v4f f = *(const v4f*)(x + 4 * i);
    v4bf o;
#pragma unroll
    for (int j = 0; j < 4; ++j) o[j] = (__bf16)f[j];
    *(v4bf*)(xbf + 4 * i) = o;
  }
}

// Re-initialize barrier + hidden-state buffers every launch (graph-replay safe).
__global__ void rnn_init_ws(unsigned* __restrict__ bar,
                            __bf16* __restrict__ h0,
                            __bf16* __restrict__ h1) {
  const size_t i = (size_t)blockIdx.x * blockDim.x + threadIdx.x;
  if (i == 0) *bar = 0u;
  const size_t n = 2u * RNN_B * RNN_D;             // per parity-buffer pair
  const size_t stride = (size_t)gridDim.x * blockDim.x;
  for (size_t j = i; j < n; j += stride) {
    h0[j] = (__bf16)0.0f;
    h1[j] = (__bf16)0.0f;
  }
}

extern "C" void kernel_launch(void* const* d_in, const int* in_sizes, int n_in,
                              void* d_out, int out_size, void* d_ws,
                              size_t ws_size, hipStream_t stream) {
  (void)in_sizes; (void)n_in; (void)out_size;
  const float* x    = (const float*)d_in[0];
  const float* Wh   = (const float*)d_in[1];
  const float* Uh   = (const float*)d_in[2];
  const float* bias = (const float*)d_in[3];
  float* out = (float*)d_out;

  char* ws = (char*)d_ws;
  unsigned* bar = (unsigned*)ws;                       // 256B reserved
  __bf16* h0 = (__bf16*)(ws + 256);                    // [2][B][D]
  __bf16* h1 = h0 + 2 * RNN_B * RNN_D;                 // [2][B][D]

  const size_t state_bytes = 256 + 2ull * (2ull * RNN_B * RNN_D) * sizeof(__bf16);
  const size_t xbf_bytes   = (size_t)RNN_B * RNN_S * RNN_D * sizeof(__bf16);
  const int use_xbf = (ws_size >= state_bytes + xbf_bytes) ? 1 : 0;
  __bf16* xbf = (__bf16*)(ws + state_bytes);

  rnn_init_ws<<<64, 256, 0, stream>>>(bar, h0, h1);
  if (use_xbf) {
    const size_t n4 = (size_t)RNN_B * RNN_S * RNN_D / 4;
    rnn_cvt_x<<<4096, 256, 0, stream>>>(x, xbf, n4);
  }

  const size_t lds_bytes = (size_t)LDS_ELEMS * sizeof(__bf16);  // ~128.5 KB
  hipFuncSetAttribute((const void*)rnn_persistent,
                      hipFuncAttributeMaxDynamicSharedMemorySize,
                      (int)lds_bytes);
  rnn_persistent<<<NWG, TPB, lds_bytes, stream>>>(x, xbf, use_xbf, Wh, Uh,
                                                  bias, out, h0, h1, bar);
}
